// SparsePooling_24257975288243
// MI455X (gfx1250) — matrix operand order repaired and checked by reading, not compiled, not run on hardware
//
#include <hip/hip_runtime.h>
#include <hip/hip_bf16.h>

typedef __attribute__((ext_vector_type(16))) _Float16 v16h;
typedef __attribute__((ext_vector_type(8)))  float    v8f;

#define B_TOK 8192
#define D_IN  512
#define H_DIM 1024
#define O_DIM 512
#define E_NUM 16
#define K_TOP 3

#define TILE_M 64
#define HC     64     // H-chunk per fused iteration
#define XLD    520    // padded strides (avoid 64-bank conflicts)
#define WLD    520    // W1^T rows [h][d]
#define W2LD   72     // W2^T rows [o][h]
#define HLD    72     // H tile rows [m][h]

// ---- WMMA fragment loaders (CDNA5 wave32 16-bit layouts, ISA 7.12.2) ----
// A 16x32 (MxK): lane m=lane&15, g=lane>>4; vgpr i holds K = (i>=4?16:0)+8g+(i&3)*2, +1
__device__ __forceinline__ v16h ld_A(const _Float16* p, int ld, int m, int g) {
  union { v16h v; unsigned u[8]; } r;
  const _Float16* q = p + m * ld + (g << 3);
#pragma unroll
  for (int i = 0; i < 8; ++i) {
    int k0 = ((i & 4) << 2) + ((i & 3) << 1);
    r.u[i] = *(const unsigned*)(q + k0);
  }
  return r.v;
}
// B 32x16 (KxN): lane n=lane&15, g=lane>>4; vgpr i holds K = (i>=4?16:0)+8g+(i&3)*2, +1
// p points at row-major [n][k] (i.e. B transposed) storage.
__device__ __forceinline__ v16h ld_B(const _Float16* p, int ld, int n, int g) {
  union { v16h v; unsigned u[8]; } r;
  const _Float16* q = p + n * ld + (g << 3);
#pragma unroll
  for (int i = 0; i < 8; ++i) r.u[i] = *(const unsigned*)(q + ((i & 4) << 2) + ((i & 3) << 1));
  return r.v;
}

// ---------------- kernel 1: zero the output (it gets scatter-accumulated) ---
__global__ void zero_out_kernel(float* __restrict__ out, int n) {
  int i = blockIdx.x * blockDim.x + threadIdx.x;
  if (i < n) out[i] = 0.0f;
}

// ---------------- kernel 2: gate logits + top-3 + softmax ------------------
__global__ void gate_topk_kernel(const float* __restrict__ X,
                                 const float* __restrict__ gW,
                                 const float* __restrict__ gb,
                                 int* __restrict__ topk_idx,
                                 float* __restrict__ topk_w) {
  int b = blockIdx.x * blockDim.x + threadIdx.x;
  if (b >= B_TOK) return;
  float acc[E_NUM];
#pragma unroll
  for (int e = 0; e < E_NUM; ++e) acc[e] = gb[e];
  const float* xr = X + (size_t)b * D_IN;
  for (int d = 0; d < D_IN; ++d) {
    float xv = xr[d];
    const float* wr = gW + d * E_NUM;   // uniform address -> scalar loads
#pragma unroll
    for (int e = 0; e < E_NUM; ++e) acc[e] = fmaf(xv, wr[e], acc[e]);
  }
  int   idx[K_TOP];
  float val[K_TOP];
#pragma unroll
  for (int j = 0; j < K_TOP; ++j) {
    int bi = 0; float bv = -1e30f;
#pragma unroll
    for (int e = 0; e < E_NUM; ++e) {
      bool taken = false;
#pragma unroll
      for (int jj = 0; jj < K_TOP; ++jj)
        if (jj < j) taken |= (idx[jj] == e);
      if (!taken && acc[e] > bv) { bv = acc[e]; bi = e; }
    }
    idx[j] = bi; val[j] = bv;
  }
  float m = val[0];                       // val[0] is the max
  float p[K_TOP], s = 0.0f;
#pragma unroll
  for (int j = 0; j < K_TOP; ++j) { p[j] = __expf(val[j] - m); s += p[j]; }
  float inv = 1.0f / s;
#pragma unroll
  for (int j = 0; j < K_TOP; ++j) {
    topk_idx[b * K_TOP + j] = idx[j];
    topk_w [b * K_TOP + j] = p[j] * inv;
  }
}

// --------- kernel 3: deterministic per-expert token compaction -------------
__global__ void route_scan_kernel(const int* __restrict__ topk_idx,
                                  const float* __restrict__ topk_w,
                                  int* __restrict__ counts,
                                  int* __restrict__ tokList,
                                  float* __restrict__ wList) {
  const int e = blockIdx.x;
  const int t = threadIdx.x;
  __shared__ int sflag[256];
  __shared__ int sbase;
  if (t == 0) sbase = 0;
  __syncthreads();
  for (int base = 0; base < B_TOK; base += 256) {
    int b = base + t;
    float wt = 0.0f; int match = 0;
#pragma unroll
    for (int j = 0; j < K_TOP; ++j)
      if (topk_idx[b * K_TOP + j] == e) { match = 1; wt = topk_w[b * K_TOP + j]; }
    sflag[t] = match;
    __syncthreads();
    // inclusive Hillis-Steele scan
    for (int off = 1; off < 256; off <<= 1) {
      int v = (t >= off) ? sflag[t - off] : 0;
      __syncthreads();
      sflag[t] += v;
      __syncthreads();
    }
    if (match) {
      int pos = sbase + sflag[t] - 1;
      tokList[e * B_TOK + pos] = b;
      wList [e * B_TOK + pos] = wt;
    }
    __syncthreads();
    if (t == 0) sbase += sflag[255];
    __syncthreads();
  }
  if (t == 0) counts[e] = sbase;
}

// --------- kernel 4: fused per-expert FFN (WMMA) + weighted scatter --------
__global__ __launch_bounds__(512, 1)
void moe_ffn_kernel(const float* __restrict__ X,
                    const float* __restrict__ W1, const float* __restrict__ b1,
                    const float* __restrict__ W2, const float* __restrict__ b2,
                    const int* __restrict__ counts,
                    const int* __restrict__ tokList,
                    const float* __restrict__ wList,
                    float* __restrict__ out) {
  const int e   = blockIdx.y;
  const int m0  = blockIdx.x * TILE_M;
  const int n_e = counts[e];
  if (m0 >= n_e) return;                       // uniform per block
  const int nrows = (n_e - m0 < TILE_M) ? (n_e - m0) : TILE_M;

  extern __shared__ char smem[];
  _Float16* Xs  = (_Float16*)smem;             // [64][520]  X tile (f16)
  _Float16* Wt  = Xs + TILE_M * XLD;           // shared W1^T / W2^T buffer
  _Float16* Hs  = Wt + 512 * W2LD;             // [64][72]   hidden chunk (f16)
  int*   tokL   = (int*)(Hs + TILE_M * HLD);
  float* twtL   = (float*)(tokL + TILE_M);

  const int t    = threadIdx.x;
  const int lane = t & 31;
  const int w    = t >> 5;                     // wave 0..15
  const int ml   = lane & 15;
  const int g    = (lane >> 4) & 1;
  const int mi   = w & 3, ni = w >> 2;         // phase-b H-tile per wave
  const int ostart = w * 32;                   // phase-d output stripe

  if (t < TILE_M) {
    int p = m0 + t;
    bool v = p < n_e;
    tokL[t] = v ? tokList[e * B_TOK + p] : 0;
    twtL[t] = v ? wList [e * B_TOK + p] : 0.0f;
  }
  __syncthreads();

  // gather + f32->f16 convert X tile
  for (int idx = t; idx < TILE_M * D_IN; idx += 512) {
    int row = idx >> 9, col = idx & 511;
    float vv = (row < nrows) ? X[(size_t)tokL[row] * D_IN + col] : 0.0f;
    Xs[row * XLD + col] = (_Float16)vv;
  }

  v8f acc[8];
  v8f zero = {};
#pragma unroll
  for (int i = 0; i < 8; ++i) acc[i] = zero;

#pragma unroll 1
  for (int hc = 0; hc < H_DIM; hc += HC) {
    __syncthreads();                           // prior phase-d done with Wt/Hs
    // W1 chunk, transposed -> [h][d]
    for (int idx = t; idx < D_IN * HC; idx += 512) {
      int d = idx >> 6, h = idx & 63;
      Wt[h * WLD + d] =
          (_Float16)W1[(size_t)e * D_IN * H_DIM + (size_t)d * H_DIM + hc + h];
    }
    __syncthreads();

    // phase b: Hc = relu(X @ W1chunk + b1); one 16x16 tile per wave
    {
      v8f c = zero;
      const _Float16* Ab = Xs + (mi * 16) * XLD;
      const _Float16* Bb = Wt + (ni * 16) * WLD;
#pragma unroll 4
      for (int kk = 0; kk < D_IN; kk += 32) {
        v16h a  = ld_A(Ab + kk, XLD, ml, g);
        v16h bb = ld_B(Bb + kk, WLD, ml, g);
        c = __builtin_amdgcn_wmma_f32_16x16x32_f16(false, a, false, bb,
                                                   (short)0, c, false, false);
      }
      float b1v = b1[e * H_DIM + hc + ni * 16 + ml];
#pragma unroll
      for (int r = 0; r < 8; ++r) {
        float hv = c[r] + b1v;
        hv = hv > 0.0f ? hv : 0.0f;
        Hs[(mi * 16 + r + (g << 3)) * HLD + ni * 16 + ml] = (_Float16)hv;
      }
    }
    __syncthreads();

    // W2 chunk, transposed -> [o][h], reusing Wt
    for (int idx = t; idx < HC * O_DIM; idx += 512) {
      int h = idx >> 9, o = idx & 511;
      Wt[o * W2LD + h] =
          (_Float16)W2[(size_t)e * H_DIM * O_DIM + (size_t)(hc + h) * O_DIM + o];
    }
    __syncthreads();

    // phase d: acc[64 x 32-stripe] += Hc @ W2chunk
#pragma unroll
    for (int kk = 0; kk < HC; kk += 32) {
      v16h a0 = ld_A(Hs + 0 * 16 * HLD + kk, HLD, ml, g);
      v16h a1 = ld_A(Hs + 1 * 16 * HLD + kk, HLD, ml, g);
      v16h a2 = ld_A(Hs + 2 * 16 * HLD + kk, HLD, ml, g);
      v16h a3 = ld_A(Hs + 3 * 16 * HLD + kk, HLD, ml, g);
#pragma unroll
      for (int nj = 0; nj < 2; ++nj) {
        v16h bb = ld_B(Wt + (ostart + nj * 16) * W2LD + kk, W2LD, ml, g);
        acc[0 * 2 + nj] = __builtin_amdgcn_wmma_f32_16x16x32_f16(
            false, a0, false, bb, (short)0, acc[0 * 2 + nj], false, false);
        acc[1 * 2 + nj] = __builtin_amdgcn_wmma_f32_16x16x32_f16(
            false, a1, false, bb, (short)0, acc[1 * 2 + nj], false, false);
        acc[2 * 2 + nj] = __builtin_amdgcn_wmma_f32_16x16x32_f16(
            false, a2, false, bb, (short)0, acc[2 * 2 + nj], false, false);
        acc[3 * 2 + nj] = __builtin_amdgcn_wmma_f32_16x16x32_f16(
            false, a3, false, bb, (short)0, acc[3 * 2 + nj], false, false);
      }
    }
  }

  // epilogue: + b2, * gate weight, scatter-accumulate into out
#pragma unroll
  for (int mi2 = 0; mi2 < 4; ++mi2) {
#pragma unroll
    for (int nj = 0; nj < 2; ++nj) {
      int o = ostart + nj * 16 + ml;
      float b2v = b2[e * O_DIM + o];
      v8f cc = acc[mi2 * 2 + nj];
#pragma unroll
      for (int r = 0; r < 8; ++r) {
        int m = mi2 * 16 + r + (g << 3);
        if (m < nrows) {
          float y = cc[r] + b2v;
          atomicAdd(out + (size_t)tokL[m] * O_DIM + o, twtL[m] * y);
        }
      }
    }
  }
}

extern "C" void kernel_launch(void* const* d_in, const int* in_sizes, int n_in,
                              void* d_out, int out_size, void* d_ws, size_t ws_size,
                              hipStream_t stream) {
  (void)in_sizes; (void)n_in; (void)out_size; (void)ws_size;
  const float* X  = (const float*)d_in[0];
  const float* gW = (const float*)d_in[1];
  const float* gb = (const float*)d_in[2];
  const float* W1 = (const float*)d_in[3];
  const float* b1 = (const float*)d_in[4];
  const float* W2 = (const float*)d_in[5];
  const float* b2 = (const float*)d_in[6];
  float* out = (float*)d_out;

  char* ws = (char*)d_ws;
  int*   topk_idx = (int*)ws;    ws += (size_t)B_TOK * K_TOP * sizeof(int);
  float* topk_w   = (float*)ws;  ws += (size_t)B_TOK * K_TOP * sizeof(float);
  int*   counts   = (int*)ws;    ws += 64;
  int*   tokList  = (int*)ws;    ws += (size_t)E_NUM * B_TOK * sizeof(int);
  float* wList    = (float*)ws;

  zero_out_kernel<<<(B_TOK * O_DIM + 255) / 256, 256, 0, stream>>>(out, B_TOK * O_DIM);
  gate_topk_kernel<<<B_TOK / 128, 128, 0, stream>>>(X, gW, gb, topk_idx, topk_w);
  route_scan_kernel<<<E_NUM, 256, 0, stream>>>(topk_idx, topk_w, counts, tokList, wList);

  size_t smem = (size_t)(TILE_M * XLD + 512 * W2LD + TILE_M * HLD) * sizeof(_Float16)
              + TILE_M * (sizeof(int) + sizeof(float));
  dim3 grid(B_TOK / TILE_M, E_NUM);
  moe_ffn_kernel<<<grid, 512, smem, stream>>>(X, W1, b1, W2, b2,
                                              counts, tokList, wList, out);
}
